// LSTMCell_71176198029872
// MI455X (gfx1250) — matrix operand order repaired and checked by reading, not compiled, not run on hardware
//
#include <hip/hip_runtime.h>
#include <stdint.h>

#define DH      4096      // hidden/input dim
#define NBATCH  1024      // batch
#define BM      64        // rows per block
#define BN      64        // per-gate cols per block
#define BK      32        // K per step
#define KSTEPS  256       // 128 (x@W) + 128 (h@U)

typedef __attribute__((ext_vector_type(16))) __bf16   v16bf;
typedef __attribute__((ext_vector_type(2)))  __bf16   v2bf;
typedef __attribute__((ext_vector_type(8)))  float    v8f;
typedef __attribute__((ext_vector_type(4)))  uint32_t v4u;
typedef __attribute__((ext_vector_type(8)))  uint32_t v8u;

// fp32 pair -> packed bf16 dword; native v_cvt_pk_bf16_f32 on gfx1250
static __device__ __forceinline__ uint32_t pack2_bf16(float a, float b) {
#if __has_builtin(__builtin_amdgcn_cvt_pk_bf16_f32)
  v2bf p = __builtin_amdgcn_cvt_pk_bf16_f32(a, b);
  return __builtin_bit_cast(uint32_t, p);
#else
  __bf16 lo = (__bf16)a;
  __bf16 hi = (__bf16)b;
  return (uint32_t)__builtin_bit_cast(uint16_t, lo) |
         ((uint32_t)__builtin_bit_cast(uint16_t, hi) << 16);
#endif
}

// branch-free activations (v_exp_f32 + v_rcp_f32; saturate correctly at +/-inf)
static __device__ __forceinline__ float sigmoid_fast(float x) {
  return __fdividef(1.0f, 1.0f + __expf(-x));
}
static __device__ __forceinline__ float tanh_fast(float x) {
  return 1.0f - __fdividef(2.0f, __expf(2.0f * x) + 1.0f);
}

__global__ __launch_bounds__(256)
void lstm_cell_wmma(const float* __restrict__ x,  const float* __restrict__ hp,
                    const float* __restrict__ cp,
                    const float* __restrict__ Wi, const float* __restrict__ Ui, const float* __restrict__ bi,
                    const float* __restrict__ Wf, const float* __restrict__ Uf, const float* __restrict__ bfv,
                    const float* __restrict__ Wg, const float* __restrict__ Ug, const float* __restrict__ bg,
                    const float* __restrict__ Wo, const float* __restrict__ Uo, const float* __restrict__ bo,
                    float* __restrict__ h_out, float* __restrict__ c_out)
{
  // bf16 tiles stored as dword (k-pair) units. A: [m][k] row-major (row = 16 dwords).
  // B per gate: transposed [n][k] (row = 16 dwords) so fragments are contiguous.
  __shared__ __align__(64) uint32_t As[2][BM * (BK / 2)];
  __shared__ __align__(64) uint32_t Bs[2][4][BN * (BK / 2)];

  const int tid  = threadIdx.x;
  const int lane = tid & 31;
  const int wave = tid >> 5;
  const int wm   = wave >> 2;   // 0..1 : which 32-row slab
  const int wn   = wave & 3;    // 0..3 : which 16-col slab
  const int m0   = blockIdx.y * BM;
  const int n0   = blockIdx.x * BN;
  const int half = lane >> 4;   // lane group (K-half select)
  const int l15  = lane & 15;

  const float* const Wp[4] = { Wi, Wf, Wg, Wo };
  const float* const Up[4] = { Ui, Uf, Ug, Uo };

  v8f acc[4][2];
  {
    v8f vz = {};
    #pragma unroll
    for (int g = 0; g < 4; ++g) { acc[g][0] = vz; acc[g][1] = vz; }
  }

  // ---- global->register prefetch staging (already bf16-packed) ----
  uint32_t a_pk[4];
  uint32_t b_pk[4][4];

  // A mapping: tokens t = tid, tid+256 ; row = t>>3 (0..63), k-quad = (t&7)*4
  const int ar  = tid >> 3;
  const int ac4 = (tid & 7) * 4;
  // B mapping: thread handles k-pair (tid>>4) of column quad (tid&15)*4, per gate.
  // Two float4 loads at rows k, k+1 -> column-wise packs -> [n][k] LDS dwords.
  const int bkp = tid >> 4;        // k-pair index 0..15
  const int bn4 = (tid & 15) * 4;  // column quad 0,4,...,60

  auto load_tiles = [&](int kk) {
    const bool ph0 = (kk < 128);
    const int  kg  = (kk & 127) * BK;
    const float* Asrc = ph0 ? x : hp;
    const float4 av0 = *(const float4*)(Asrc + (size_t)(m0 + ar)      * DH + kg + ac4);
    const float4 av1 = *(const float4*)(Asrc + (size_t)(m0 + 32 + ar) * DH + kg + ac4);
    a_pk[0] = pack2_bf16(av0.x, av0.y);
    a_pk[1] = pack2_bf16(av0.z, av0.w);
    a_pk[2] = pack2_bf16(av1.x, av1.y);
    a_pk[3] = pack2_bf16(av1.z, av1.w);
    #pragma unroll
    for (int g = 0; g < 4; ++g) {
      const float* Wsrc = ph0 ? Wp[g] : Up[g];
      const float4 w0 = *(const float4*)(Wsrc + (size_t)(kg + 2 * bkp    ) * DH + n0 + bn4);
      const float4 w1 = *(const float4*)(Wsrc + (size_t)(kg + 2 * bkp + 1) * DH + n0 + bn4);
      b_pk[g][0] = pack2_bf16(w0.x, w1.x);
      b_pk[g][1] = pack2_bf16(w0.y, w1.y);
      b_pk[g][2] = pack2_bf16(w0.z, w1.z);
      b_pk[g][3] = pack2_bf16(w0.w, w1.w);
    }
  };

  auto store_tiles = [&](int buf) {
    const int aq = ac4 >> 1;
    As[buf][ ar       * 16 + aq    ] = a_pk[0];
    As[buf][ ar       * 16 + aq + 1] = a_pk[1];
    As[buf][(ar + 32) * 16 + aq    ] = a_pk[2];
    As[buf][(ar + 32) * 16 + aq + 1] = a_pk[3];
    #pragma unroll
    for (int g = 0; g < 4; ++g) {
      #pragma unroll
      for (int j = 0; j < 4; ++j)
        Bs[buf][g][(bn4 + j) * 16 + bkp] = b_pk[g][j];
    }
  };

  auto compute = [&](int buf) {
    // A fragment (ISA 16-bit 16x32 layout): lanes 0-15 hold K{0..7,16..23},
    // lanes 16-31 hold K{8..15,24..31}; two b128 chunks at +4*half and +8+4*half.
    v16bf afrag[2];
    #pragma unroll
    for (int mt = 0; mt < 2; ++mt) {
      const uint32_t* ap = &As[buf][(wm * 32 + mt * 16 + l15) * 16 + half * 4];
      v4u q0 = *(const v4u*)(ap);
      v4u q1 = *(const v4u*)(ap + 8);
      v8u w  = __builtin_shufflevector(q0, q1, 0, 1, 2, 3, 4, 5, 6, 7);
      afrag[mt] = __builtin_bit_cast(v16bf, w);
    }
    #pragma unroll
    for (int g = 0; g < 4; ++g) {
      // B fragment: lane -> column (wn*16 + l15), same K order as A (K{0..7,16..23} / K{8..15,24..31}).
      const uint32_t* bp = &Bs[buf][g][(wn * 16 + l15) * 16 + half * 4];
      v4u bq0 = *(const v4u*)(bp);
      v4u bq1 = *(const v4u*)(bp + 8);
      v16bf bfrag = __builtin_bit_cast(v16bf, __builtin_shufflevector(bq0, bq1, 0, 1, 2, 3, 4, 5, 6, 7));
      #pragma unroll
      for (int mt = 0; mt < 2; ++mt) {
        acc[g][mt] = __builtin_amdgcn_wmma_f32_16x16x32_bf16(
            false, afrag[mt], false, bfrag, (short)0, acc[g][mt], false, false);
      }
    }
  };

  // ---- pipelined main loop: double-buffered LDS, one barrier per K-step ----
  load_tiles(0);
  store_tiles(0);
  __syncthreads();

  for (int kk = 0; kk < KSTEPS; ++kk) {
    const int buf = kk & 1;
    if (kk + 1 < KSTEPS) load_tiles(kk + 1);
    compute(buf);
    if (kk + 1 < KSTEPS) {
      store_tiles(buf ^ 1);
      __syncthreads();
    }
  }

  // ---- epilogue: gates + cell update, write h and c ----
  // C/D layout: VGPR i -> row (base + i + 8*half), col = base + (lane&15)
  const int col = n0 + wn * 16 + l15;
  const float bvi = bi[col], bvf = bfv[col], bvg = bg[col], bvo = bo[col];
  #pragma unroll
  for (int mt = 0; mt < 2; ++mt) {
    #pragma unroll
    for (int i = 0; i < 8; ++i) {
      const int row = m0 + wm * 32 + mt * 16 + half * 8 + i;
      const size_t idx = (size_t)row * DH + col;
      const float zi = acc[0][mt][i] + bvi;
      const float zf = acc[1][mt][i] + bvf;
      const float zg = acc[2][mt][i] + bvg;
      const float zo = acc[3][mt][i] + bvo;
      const float ig = sigmoid_fast(zi);
      const float fg = sigmoid_fast(zf);
      const float gg = tanh_fast(zg);
      const float og = sigmoid_fast(zo);
      const float cv = fg * cp[idx] + ig * gg;
      c_out[idx] = cv;
      h_out[idx] = og * tanh_fast(cv);
    }
  }
}

extern "C" void kernel_launch(void* const* d_in, const int* in_sizes, int n_in,
                              void* d_out, int out_size, void* d_ws, size_t ws_size,
                              hipStream_t stream) {
  (void)in_sizes; (void)n_in; (void)out_size; (void)d_ws; (void)ws_size;
  const float* x  = (const float*)d_in[0];
  const float* hp = (const float*)d_in[1];
  const float* cp = (const float*)d_in[2];
  const float* Wi = (const float*)d_in[3];
  const float* Ui = (const float*)d_in[4];
  const float* bi = (const float*)d_in[5];
  const float* Wf = (const float*)d_in[6];
  const float* Uf = (const float*)d_in[7];
  const float* bf = (const float*)d_in[8];
  const float* Wg = (const float*)d_in[9];
  const float* Ug = (const float*)d_in[10];
  const float* bg = (const float*)d_in[11];
  const float* Wo = (const float*)d_in[12];
  const float* Uo = (const float*)d_in[13];
  const float* bo = (const float*)d_in[14];

  float* h_out = (float*)d_out;
  float* c_out = h_out + (size_t)NBATCH * DH;

  dim3 grid(DH / BN, NBATCH / BM);   // (64, 16)
  dim3 block(256);
  lstm_cell_wmma<<<grid, block, 0, stream>>>(
      x, hp, cp, Wi, Ui, bi, Wf, Uf, bf, Wg, Ug, bg, Wo, Uo, bo, h_out, c_out);
}
